// TransformerEncoderLayer_49117245997327
// MI455X (gfx1250) — matrix operand order repaired and checked by reading, not compiled, not run on hardware
//
#include <hip/hip_runtime.h>
#include <stdint.h>

// ---------------------------------------------------------------------------
// CDNA5 (gfx1250) transformer encoder layer, bf16 WMMA everywhere.
//  - All WMMA operands are contiguous 16B (b128) loads (global or LDS).
//  - Weights kept TRANSPOSED in global ([N][K]); V produced transposed.
//  - Attention stages K/V tiles into LDS once per block via async-to-LDS
//    copies (ASYNCcnt), double-buffered to overlap copy with WMMA.
// ---------------------------------------------------------------------------

typedef __attribute__((ext_vector_type(16))) __bf16        v16bf;
typedef __attribute__((ext_vector_type(8)))  float         v8f;
typedef __attribute__((ext_vector_type(4)))  unsigned int  u32x4;

union Frag16 {                 // 32 bytes: one wmma A or B operand (bf16)
    u32x4          q[2];
    unsigned short u[16];
    v16bf          v;
};
union FragC {                  // 32 bytes: one wmma C/D accumulator (f32)
    v8f   v;
    float f[8];
};

static __device__ __forceinline__ unsigned short f2bf(float f) {
    union { float f; unsigned int u; } c; c.f = f;
    unsigned int u = c.u;
    return (unsigned short)((u + 0x7FFFu + ((u >> 16) & 1u)) >> 16);
}

static __device__ __forceinline__ v8f wmma_bf16(v16bf a, v16bf b, v8f c) {
    return __builtin_amdgcn_wmma_f32_16x16x32_bf16(
        /*neg_a=*/false, a, /*neg_b=*/false, b,
        /*c_mod=*/(short)0, c, /*reuse_a=*/false, /*reuse_b=*/false);
}

// --- CDNA5 async global->LDS copy (ASYNCcnt), per-lane 16B ------------------
// LDS aperture maps addr[31:0] directly to the LDS offset (ISA 10.2), so the
// low 32 bits of the generic shared pointer are the VDST LDS address.
static __device__ __forceinline__ void async_copy_b128(const void* g, void* l) {
    unsigned loff = (unsigned)(uintptr_t)l;
    unsigned long long ga = (unsigned long long)(uintptr_t)g;
    asm volatile("global_load_async_to_lds_b128 %0, %1, off"
                 :: "v"(loff), "v"(ga) : "memory");
}
static __device__ __forceinline__ void wait_async_le2() {
    asm volatile("s_wait_asynccnt 0x2" ::: "memory");
}
static __device__ __forceinline__ void wait_async_0() {
    asm volatile("s_wait_asynccnt 0x0" ::: "memory");
}

// ---------------------------------------------------------------------------
// fp32 -> bf16 conversion (layout-preserving)
// ---------------------------------------------------------------------------
__global__ __launch_bounds__(256)
void cvt_f32_bf16(const float* __restrict__ in, unsigned short* __restrict__ out,
                  size_t n) {
    size_t i = (size_t)blockIdx.x * blockDim.x + threadIdx.x;
    size_t stride = (size_t)gridDim.x * blockDim.x;
    for (; i < n; i += stride) out[i] = f2bf(in[i]);
}

// ---------------------------------------------------------------------------
// fp32 [Kd][Nd] -> bf16 [Nd][Kd] (convert + transpose, 32x32 LDS tile)
// ---------------------------------------------------------------------------
__global__ __launch_bounds__(256)
void cvt_transpose_f32_bf16(const float* __restrict__ in,
                            unsigned short* __restrict__ out,
                            int Kd, int Nd) {
    __shared__ float tile[32][33];
    const int k0 = blockIdx.y * 32, n0 = blockIdx.x * 32;
    const int r  = threadIdx.x >> 3;
    const int c4 = (threadIdx.x & 7) * 4;
    const float* src = in + (size_t)(k0 + r) * Nd + n0 + c4;
#pragma unroll
    for (int i = 0; i < 4; ++i) tile[r][c4 + i] = src[i];
    __syncthreads();
    unsigned short* dst = out + (size_t)(n0 + r) * Kd + k0 + c4;
#pragma unroll
    for (int i = 0; i < 4; ++i) dst[i] = f2bf(tile[c4 + i][r]);
}

// ---------------------------------------------------------------------------
// bf16 GEMM, B transposed:  C[M,N] = A[M,K] @ Bt[N,K]^T
// Block: 256 threads = 8 waves; tile M=128, N=64; wave = 32x32 (2x2 frags).
// No LDS: all frags are contiguous global b128 pairs; weights stay L2-hot.
// ---------------------------------------------------------------------------
#define GBM 128
#define GBN 64
#define GBK 32

__global__ __launch_bounds__(256)
void gemm_bf16_tn(const unsigned short* __restrict__ A,   // [M][K] bf16
                  const unsigned short* __restrict__ Bt,  // [N][K] bf16
                  const float* __restrict__ bias,
                  const float* __restrict__ residual,
                  float* __restrict__ Cf,
                  unsigned short* __restrict__ Cb,        // [M][N] bf16
                  unsigned short* __restrict__ CbT,       // [N][M] bf16
                  int M, int N, int K, int relu) {
    const int lane = threadIdx.x & 31;
    const int w    = threadIdx.x >> 5;
    const int wm   = w & 3;
    const int wn   = w >> 2;
    const int hlf  = lane >> 4;
    const int l16  = lane & 15;

    const int M0 = blockIdx.y * GBM;
    const int N0 = blockIdx.x * GBN;

    const unsigned short* arow[2];
#pragma unroll
    for (int fm = 0; fm < 2; ++fm)
        arow[fm] = A + (size_t)(M0 + wm * 32 + fm * 16 + l16) * K + hlf * 8;
    const unsigned short* brow[2];
#pragma unroll
    for (int fn = 0; fn < 2; ++fn)
        brow[fn] = Bt + (size_t)(N0 + wn * 32 + fn * 16 + l16) * K + hlf * 8;

    FragC acc[2][2];
#pragma unroll
    for (int i = 0; i < 2; ++i)
#pragma unroll
        for (int j = 0; j < 2; ++j)
#pragma unroll
            for (int r = 0; r < 8; ++r) acc[i][j].f[r] = 0.f;

    for (int k0 = 0; k0 < K; k0 += GBK) {
        Frag16 af[2], bf[2];
#pragma unroll
        for (int fm = 0; fm < 2; ++fm) {
            af[fm].q[0] = *(const u32x4*)(arow[fm] + k0);
            af[fm].q[1] = *(const u32x4*)(arow[fm] + k0 + 16);
        }
#pragma unroll
        for (int fn = 0; fn < 2; ++fn) {
            bf[fn].q[0] = *(const u32x4*)(brow[fn] + k0);
            bf[fn].q[1] = *(const u32x4*)(brow[fn] + k0 + 16);
        }
        if (k0 + GBK < K) {
            __builtin_prefetch(arow[0] + k0 + GBK, 0, 3);
            __builtin_prefetch(brow[0] + k0 + GBK, 0, 3);
        }
#pragma unroll
        for (int fn = 0; fn < 2; ++fn)
#pragma unroll
            for (int fm = 0; fm < 2; ++fm)
                acc[fm][fn].v = wmma_bf16(af[fm].v, bf[fn].v, acc[fm][fn].v);
    }

    // epilogue — C layout: N = lane%16, M = r + 8*(lane/16)
#pragma unroll
    for (int fm = 0; fm < 2; ++fm)
#pragma unroll
        for (int fn = 0; fn < 2; ++fn) {
            const int col = N0 + wn * 32 + fn * 16 + l16;
            const float bs = bias ? bias[col] : 0.f;
#pragma unroll
            for (int r = 0; r < 8; ++r) {
                const int row = M0 + wm * 32 + fm * 16 + r + 8 * hlf;
                float v = acc[fm][fn].f[r] + bs;
                if (residual) v += residual[(size_t)row * N + col];
                if (relu) v = v > 0.f ? v : 0.f;
                if (Cf)  Cf[(size_t)row * N + col] = v;
                if (Cb)  Cb[(size_t)row * N + col] = f2bf(v);
                if (CbT) CbT[(size_t)col * M + row] = f2bf(v);
            }
        }
}

// ---------------------------------------------------------------------------
// Fused attention (flash style, online softmax). dk=64, scale 1/8.
// Q,K bf16 [T][1024]; V bf16 TRANSPOSED [1024][T]. Block = 8 waves = 128 q
// rows of one (b,h); wave = 16 q rows; kv step 32.
// K/V tiles (4KB each) staged ONCE per block into LDS via async-to-LDS b128
// copies, double-buffered: copy of tile i+1 overlaps WMMA on tile i.
// ---------------------------------------------------------------------------
__global__ __launch_bounds__(256)
void attention_fused(const unsigned short* __restrict__ Q,
                     const unsigned short* __restrict__ Kx,
                     const unsigned short* __restrict__ Vt,
                     unsigned short* __restrict__ ctx,
                     int S, int T) {
    __shared__ __align__(16) unsigned short lds_k[2][32][72]; // 32 kv x 64 dk
    __shared__ __align__(16) unsigned short lds_v[2][64][40]; // 64 dk x 32 kv
    __shared__ __align__(16) unsigned short lds_p[8][16][40]; // per-wave P

    const int tid  = threadIdx.x;
    const int lane = tid & 31;
    const int w    = tid >> 5;
    const int hlf  = lane >> 4;
    const int l16  = lane & 15;

    const int blocks_per_bh = S / 128;
    const int bh   = blockIdx.x / blocks_per_bh;
    const int qblk = blockIdx.x % blocks_per_bh;
    const int b    = bh >> 4;           // NUM_HEADS = 16
    const int h    = bh & 15;
    const size_t tok0 = (size_t)b * S;
    const int hc   = h * 64;
    const int q0   = qblk * 128 + w * 16;

    // async staging: K tile 32 rows x 8 chunks, V tile 64 rows x 4 chunks
    const int krow = tid >> 3, kchk = tid & 7;
    const int vrow = tid >> 2, vchk = tid & 3;

    auto issue_tile = [&](int kv0, int bs) {
        async_copy_b128(Kx + (tok0 + kv0 + krow) * 1024 + hc + kchk * 8,
                        &lds_k[bs][krow][kchk * 8]);
        async_copy_b128(Vt + (size_t)(hc + vrow) * T + tok0 + kv0 + vchk * 8,
                        &lds_v[bs][vrow][vchk * 8]);
    };

    // Q A-frags for dk chunks [0,32) and [32,64)
    Frag16 aq[2];
    {
        const unsigned short* qp = Q + (tok0 + q0 + l16) * 1024 + hc;
#pragma unroll
        for (int c = 0; c < 2; ++c) {
            aq[c].q[0] = *(const u32x4*)(qp + c * 32 + hlf * 8);
            aq[c].q[1] = *(const u32x4*)(qp + c * 32 + 16 + hlf * 8);
        }
    }

    float mrow[8], lrow[8];
    FragC o[4];
#pragma unroll
    for (int r = 0; r < 8; ++r) { mrow[r] = -1e30f; lrow[r] = 0.f; }
#pragma unroll
    for (int f = 0; f < 4; ++f)
#pragma unroll
        for (int r = 0; r < 8; ++r) o[f].f[r] = 0.f;

    const int ntiles = S / 32;
    issue_tile(0, 0);

    for (int it = 0; it < ntiles; ++it) {
        const int bs = it & 1;
        if (it + 1 < ntiles) {
            issue_tile((it + 1) * 32, bs ^ 1);  // overlap next copy
            wait_async_le2();                   // tile `it` (2 older ops) done
        } else {
            wait_async_0();
        }
        __syncthreads();                        // staged tile visible block-wide

        // S tile: 16 q x 32 kv as two 16x16 C frags (kv-column halves)
        FragC s[2];
#pragma unroll
        for (int nh = 0; nh < 2; ++nh) {
            Frag16 bk[2];
#pragma unroll
            for (int c = 0; c < 2; ++c) {
                bk[c].q[0] = *(const u32x4*)&lds_k[bs][nh * 16 + l16][c * 32 + hlf * 8];
                bk[c].q[1] = *(const u32x4*)&lds_k[bs][nh * 16 + l16][c * 32 + 16 + hlf * 8];
            }
            v8f cc = {0.f, 0.f, 0.f, 0.f, 0.f, 0.f, 0.f, 0.f};
            cc = wmma_bf16(aq[0].v, bk[0].v, cc);
            cc = wmma_bf16(aq[1].v, bk[1].v, cc);
            s[nh].v = cc;
        }

        // online softmax; row M = r + 8*hlf lives across the 16-lane half
#pragma unroll
        for (int r = 0; r < 8; ++r) {
            const float s0 = s[0].f[r] * 0.125f;
            const float s1 = s[1].f[r] * 0.125f;
            float mx = fmaxf(s0, s1);
#pragma unroll
            for (int msk = 1; msk < 16; msk <<= 1)
                mx = fmaxf(mx, __shfl_xor(mx, msk, 32));
            const float mnew  = fmaxf(mrow[r], mx);
            const float alpha = __expf(mrow[r] - mnew);
            const float p0 = __expf(s0 - mnew);
            const float p1 = __expf(s1 - mnew);
            float ps = p0 + p1;
#pragma unroll
            for (int msk = 1; msk < 16; msk <<= 1)
                ps += __shfl_xor(ps, msk, 32);
            lrow[r] = lrow[r] * alpha + ps;
            mrow[r] = mnew;
#pragma unroll
            for (int f = 0; f < 4; ++f) o[f].f[r] *= alpha;
            const int m = r + 8 * hlf;
            lds_p[w][m][l16]      = f2bf(p0);
            lds_p[w][m][16 + l16] = f2bf(p1);
        }
        __syncthreads();   // P repack visible (uniform across all 8 waves)

        // P as A frag (16 x 32): A-layout rows contiguous -> ds b128 pair
        Frag16 pa;
        pa.q[0] = *(const u32x4*)&lds_p[w][l16][hlf * 8];
        pa.q[1] = *(const u32x4*)&lds_p[w][l16][16 + hlf * 8];

        // O += P @ V : B frag n = dk (Vt row), k = kv contiguous -> b128 pair
#pragma unroll
        for (int f = 0; f < 4; ++f) {
            Frag16 vb;
            vb.q[0] = *(const u32x4*)&lds_v[bs][f * 16 + l16][hlf * 8];
            vb.q[1] = *(const u32x4*)&lds_v[bs][f * 16 + l16][16 + hlf * 8];
            o[f].v = wmma_bf16(pa.v, vb.v, o[f].v);
        }
        __syncthreads();   // all reads of buffer `bs` done before it refills
    }

    // normalize and emit bf16 context [T][1024]
#pragma unroll
    for (int f = 0; f < 4; ++f)
#pragma unroll
        for (int r = 0; r < 8; ++r) {
            const int row = r + 8 * hlf;
            const float val = o[f].f[r] / lrow[r];
            ctx[(tok0 + q0 + row) * 1024 + hc + f * 16 + l16] = f2bf(val);
        }
}

// ---------------------------------------------------------------------------
// LayerNorm over last dim (D), one block per row. Emits f32 (+ optional bf16).
// ---------------------------------------------------------------------------
__global__ __launch_bounds__(256)
void layernorm_rows(const float* __restrict__ in, const float* __restrict__ g,
                    const float* __restrict__ bta, float* __restrict__ outf,
                    unsigned short* __restrict__ outb, int D) {
    __shared__ float red[16];
    const int row = blockIdx.x;
    const float* x = in + (size_t)row * D;

    float s = 0.f, s2 = 0.f;
    for (int i = threadIdx.x; i < D; i += blockDim.x) {
        const float v = x[i];
        s += v; s2 += v * v;
    }
#pragma unroll
    for (int m = 1; m < 32; m <<= 1) {
        s  += __shfl_xor(s,  m, 32);
        s2 += __shfl_xor(s2, m, 32);
    }
    const int w = threadIdx.x >> 5, lane = threadIdx.x & 31;
    if (lane == 0) { red[w] = s; red[8 + w] = s2; }
    __syncthreads();
    if (w == 0) {
        s  = (lane < 8) ? red[lane]     : 0.f;
        s2 = (lane < 8) ? red[8 + lane] : 0.f;
#pragma unroll
        for (int m = 1; m < 8; m <<= 1) {
            s  += __shfl_xor(s,  m, 32);
            s2 += __shfl_xor(s2, m, 32);
        }
        if (lane == 0) { red[0] = s; red[1] = s2; }
    }
    __syncthreads();
    const float mean = red[0] / (float)D;
    const float var  = red[1] / (float)D - mean * mean;
    const float inv  = rsqrtf(var + 1e-5f);
    for (int i = threadIdx.x; i < D; i += blockDim.x) {
        const float v = (x[i] - mean) * inv * g[i] + bta[i];
        if (outf) outf[(size_t)row * D + i] = v;
        if (outb) outb[(size_t)row * D + i] = f2bf(v);
    }
}

// ---------------------------------------------------------------------------
// Host-side orchestration
// ---------------------------------------------------------------------------
extern "C" void kernel_launch(void* const* d_in, const int* in_sizes, int n_in,
                              void* d_out, int out_size, void* d_ws, size_t ws_size,
                              hipStream_t stream) {
    (void)in_sizes; (void)n_in; (void)out_size; (void)ws_size;

    const float* x    = (const float*)d_in[0];
    const float* Wq   = (const float*)d_in[1];
    const float* bq   = (const float*)d_in[2];
    const float* Wk   = (const float*)d_in[3];
    const float* bk   = (const float*)d_in[4];
    const float* Wv   = (const float*)d_in[5];
    const float* bv   = (const float*)d_in[6];
    const float* Wo   = (const float*)d_in[7];
    const float* bo   = (const float*)d_in[8];
    const float* W1   = (const float*)d_in[9];
    const float* b1   = (const float*)d_in[10];
    const float* W2   = (const float*)d_in[11];
    const float* b2   = (const float*)d_in[12];
    const float* ln1g = (const float*)d_in[13];
    const float* ln1b = (const float*)d_in[14];
    const float* ln2g = (const float*)d_in[15];
    const float* ln2b = (const float*)d_in[16];
    float* out = (float*)d_out;

    const int S = 2048, D = 1024, DM = 4096;
    const int T = 2 * S;                 // 4096 tokens

    char* ws = (char*)d_ws;
    size_t off = 0;
    auto alloc = [&](size_t bytes) -> void* {
        void* p = ws + off;
        off += (bytes + 255) & ~(size_t)255;
        return p;
    };

    unsigned short* xb   = (unsigned short*)alloc((size_t)T * D * 2);
    unsigned short* Wqt  = (unsigned short*)alloc((size_t)D * D * 2);   // [N][K]
    unsigned short* Wkt  = (unsigned short*)alloc((size_t)D * D * 2);
    unsigned short* Wvt  = (unsigned short*)alloc((size_t)D * D * 2);
    unsigned short* Wot  = (unsigned short*)alloc((size_t)D * D * 2);
    unsigned short* W1t  = (unsigned short*)alloc((size_t)D * DM * 2);  // [4096][1024]
    unsigned short* W2t  = (unsigned short*)alloc((size_t)DM * D * 2);  // [1024][4096]
    unsigned short* Qb   = (unsigned short*)alloc((size_t)T * D * 2);   // [T][1024]
    unsigned short* Kb   = (unsigned short*)alloc((size_t)T * D * 2);   // [T][1024]
    unsigned short* Vbt  = (unsigned short*)alloc((size_t)T * D * 2);   // [1024][T]
    unsigned short* ctxb = (unsigned short*)alloc((size_t)T * D * 2);
    float*          y1   = (float*)alloc((size_t)T * D * 4);
    float*          x1f  = (float*)alloc((size_t)T * D * 4);
    unsigned short* x1b  = (unsigned short*)alloc((size_t)T * D * 2);
    unsigned short* hb   = (unsigned short*)alloc((size_t)T * DM * 2);
    float*          y2   = (float*)alloc((size_t)T * D * 4);

    // 1) conversions: x plain; weights convert+transpose to [N][K]
    cvt_f32_bf16<<<512, 256, 0, stream>>>(x, xb, (size_t)T * D);
    dim3 tDD(D / 32, D / 32);
    cvt_transpose_f32_bf16<<<tDD, 256, 0, stream>>>(Wq, Wqt, D, D);
    cvt_transpose_f32_bf16<<<tDD, 256, 0, stream>>>(Wk, Wkt, D, D);
    cvt_transpose_f32_bf16<<<tDD, 256, 0, stream>>>(Wv, Wvt, D, D);
    cvt_transpose_f32_bf16<<<tDD, 256, 0, stream>>>(Wo, Wot, D, D);
    dim3 tD1(DM / 32, D / 32);           // W1: [D][DM] -> [DM][D]
    cvt_transpose_f32_bf16<<<tD1, 256, 0, stream>>>(W1, W1t, D, DM);
    dim3 tD2(D / 32, DM / 32);           // W2: [DM][D] -> [D][DM]
    cvt_transpose_f32_bf16<<<tD2, 256, 0, stream>>>(W2, W2t, DM, D);

    // 2) Q, K (token-major bf16), V (transposed bf16 [1024][T])
    dim3 gD(D / GBN, T / GBM);           // 16 x 32
    gemm_bf16_tn<<<gD, 256, 0, stream>>>(xb, Wqt, bq, nullptr, nullptr, Qb, nullptr, T, D, D, 0);
    gemm_bf16_tn<<<gD, 256, 0, stream>>>(xb, Wkt, bk, nullptr, nullptr, Kb, nullptr, T, D, D, 0);
    gemm_bf16_tn<<<gD, 256, 0, stream>>>(xb, Wvt, bv, nullptr, nullptr, nullptr, Vbt, T, D, D, 0);

    // 3) fused attention: 2 batches * 16 heads * (S/128) q-blocks
    attention_fused<<<2 * 16 * (S / 128), 256, 0, stream>>>(Qb, Kb, Vbt, ctxb, S, T);

    // 4) output projection + residual(x), then LN1 -> x1 (f32 + bf16)
    gemm_bf16_tn<<<gD, 256, 0, stream>>>(ctxb, Wot, bo, x, y1, nullptr, nullptr, T, D, D, 0);
    layernorm_rows<<<T, 256, 0, stream>>>(y1, ln1g, ln1b, x1f, x1b, D);

    // 5) MLP
    dim3 gM(DM / GBN, T / GBM);          // 64 x 32
    gemm_bf16_tn<<<gM, 256, 0, stream>>>(x1b, W1t, b1, nullptr, nullptr, hb, nullptr, T, DM, D, 1);
    gemm_bf16_tn<<<gD, 256, 0, stream>>>(hb,  W2t, b2, x1f, y2, nullptr, nullptr, T, D, DM, 0);

    // 6) LN2 -> final output (f32)
    layernorm_rows<<<T, 256, 0, stream>>>(y2, ln2g, ln2b, out, nullptr, D);
}
